// MultiHeadAttention_64106681860559
// MI455X (gfx1250) — matrix operand order repaired and checked by reading, not compiled, not run on hardware
//
#include <hip/hip_runtime.h>

typedef __attribute__((ext_vector_type(16))) __bf16 v16bf;
typedef __attribute__((ext_vector_type(8)))  __bf16 v8bf;
typedef __attribute__((ext_vector_type(4)))  __bf16 v4bf;
typedef __attribute__((ext_vector_type(8)))  float  v8f;
typedef int v4i __attribute__((vector_size(16)));

// ---------------------------------------------------------------------------
// WMMA helper: D = A(16x32 bf16) * B(32x16 bf16) + C(16x16 f32)
// ---------------------------------------------------------------------------
static __device__ inline v8f wmma_bf16(v16bf a, v16bf b, v8f c) {
  return __builtin_amdgcn_wmma_f32_16x16x32_bf16(
      /*neg_a=*/false, a, /*neg_b=*/false, b,
      /*c_mod=*/(short)0, c, /*reuse_a=*/false, /*reuse_b=*/false);
}

// ---------------------------------------------------------------------------
// gfx1250 async global->LDS staging (ASYNCcnt path) with sync fallback.
// Builtin signature (from compiler diagnostic): arg0 = v4i addrspace(1)*,
// arg1 = v4i addrspace(3)*, then imm offset + imm cpol.
// ---------------------------------------------------------------------------
static __device__ inline void stage_b128(const __bf16* g, __bf16* l) {
#if __has_builtin(__builtin_amdgcn_global_load_async_to_lds_b128)
  __builtin_amdgcn_global_load_async_to_lds_b128(
      (__attribute__((address_space(1))) v4i*)g,
      (__attribute__((address_space(3))) v4i*)l, 0, 0);
#else
  *(v8bf*)l = *(const v8bf*)g;
#endif
}
static __device__ inline void stage_wait() {
#if __has_builtin(__builtin_amdgcn_global_load_async_to_lds_b128) && \
    __has_builtin(__builtin_amdgcn_s_wait_asynccnt)
  __builtin_amdgcn_s_wait_asynccnt(0);
#endif
}

// ---------------------------------------------------------------------------
// Fragment loaders (wave32 layouts per CDNA5 ISA 7.12.2). Contraction dim is
// always the fast (contiguous) dim of the source -> two 16B vector loads.
//
// A 16x32 (MxK): lane 0..15: row=lane, K={k0..+7} U {k0+16..+23}
//                lane 16..31: row=lane-16, K={k0+8..+15} U {k0+24..+31}
// ---------------------------------------------------------------------------
static __device__ inline v16bf ld_a_rowmajor(const __bf16* p, int ld, int r0,
                                             int k0, int lane) {
  int r  = r0 + (lane & 15);
  int kb = k0 + ((lane >> 4) << 3);
  const __bf16* base = p + r * ld + kb;
  union { v16bf v; v8bf h[2]; } u;
  u.h[0] = *(const v8bf*)(base);
  u.h[1] = *(const v8bf*)(base + 16);
  return u.v;
}

// B 32x16 (KxN), source is N-major row-major (B^T): lane 0..15 hold col
// n=lane with K={k0..+7} U {k0+16..+23}, lanes 16..31 hold K={k0+8..+15} U {k0+24..+31}.
static __device__ inline v16bf ld_bt_rowmajor(const __bf16* p, int ld, int c0,
                                              int k0, int lane) {
  int n  = c0 + (lane & 15);
  int kb = k0 + ((lane >> 4) << 3);
  const __bf16* base = p + n * ld + kb;
  union { v16bf v; v8bf h[2]; } u;
  u.h[0] = *(const v8bf*)(base);
  u.h[1] = *(const v8bf*)(base + 16);
  return u.v;
}

// ---------------------------------------------------------------------------
// fp32 -> bf16 conversion (vectorized x4)
// ---------------------------------------------------------------------------
__global__ void cvt_f32_bf16_kernel(const float* __restrict__ in,
                                    __bf16* __restrict__ out, int n4) {
  int i = blockIdx.x * blockDim.x + threadIdx.x;
  if (i < n4) {
    float4 f = ((const float4*)in)[i];
    v4bf o;
    o[0] = (__bf16)f.x; o[1] = (__bf16)f.y;
    o[2] = (__bf16)f.z; o[3] = (__bf16)f.w;
    ((v4bf*)out)[i] = o;
  }
}

// ---------------------------------------------------------------------------
// LDS-tiled transpose (+ convert to bf16): out[C x R] = in[R x C]^T
// block (32,8), grid (C/32, R/32)
// ---------------------------------------------------------------------------
template <typename TIN>
__global__ __launch_bounds__(256)
void transpose_to_bf16_kernel(const TIN* __restrict__ in, int ldin,
                              __bf16* __restrict__ out, int ldout) {
  __shared__ __bf16 tile[32][33];
  const int ct = blockIdx.x * 32;
  const int rt = blockIdx.y * 32;
  const int tx = threadIdx.x, ty = threadIdx.y;
#pragma unroll
  for (int i = 0; i < 32; i += 8)
    tile[ty + i][tx] = (__bf16)(float)in[(rt + ty + i) * ldin + ct + tx];
  __syncthreads();
#pragma unroll
  for (int i = 0; i < 32; i += 8)
    out[(ct + ty + i) * ldout + rt + tx] = tile[tx][ty + i];
}

// ---------------------------------------------------------------------------
// Tiled WMMA GEMM, fused bias: C[M,N] = A[M,K] @ BT[N,K]^T + bias[N]
// Block = 256 threads (8 waves). All waves share the same 16 A-rows: the
// 16 x K bf16 slab (32 KB) is staged once into LDS (async global->LDS),
// then each wave computes a 16(M) x 64(N) strip streaming B^T from L2.
// Grid = (M/16, N/512). Requires K <= 1024.
// ---------------------------------------------------------------------------
template <bool OUT_BF16>
__global__ __launch_bounds__(256)
void gemm_bias_kernel(const __bf16* __restrict__ A, const __bf16* __restrict__ BT,
                      const float* __restrict__ bias, void* __restrict__ C,
                      int K, int lda, int ldbt, int ldc) {
  __shared__ __bf16 As[16 * 1024];
  const int lane = threadIdx.x & 31;
  const int wave = threadIdx.x >> 5;
  const int row0 = blockIdx.x * 16;
  const int col0 = blockIdx.y * 512 + wave * 64;

  // cooperative stage of A rows [row0, row0+16) x [0, K)
  const int kv = K >> 3;  // 8 bf16 per b128 chunk
  for (int c = threadIdx.x; c < 16 * kv; c += 256) {
    int r  = c / kv;
    int kk = (c - r * kv) << 3;
    stage_b128(A + (row0 + r) * lda + kk, As + r * K + kk);
  }
  stage_wait();
  __syncthreads();

  v8f acc[4] = {};
  for (int k0 = 0; k0 < K; k0 += 32) {
    v16bf a = ld_a_rowmajor(As, K, 0, k0, lane);   // from LDS
#pragma unroll
    for (int j = 0; j < 4; ++j) {
      v16bf b = ld_bt_rowmajor(BT, ldbt, col0 + 16 * j, k0, lane);
      acc[j] = wmma_bf16(a, b, acc[j]);
    }
  }

  const int rloc = (lane >> 4) << 3;  // C/D layout: row = vgpr + 8*(lane>=16)
  const int cloc = lane & 15;
#pragma unroll
  for (int j = 0; j < 4; ++j) {
    int c = col0 + 16 * j + cloc;
    float bv = bias[c];
#pragma unroll
    for (int v = 0; v < 8; ++v) {
      int r = row0 + rloc + v;
      float val = acc[j][v] + bv;
      if (OUT_BF16) ((__bf16*)C)[r * ldc + c] = (__bf16)val;
      else          ((float*)C)[r * ldc + c]  = val;
    }
  }
}

// ---------------------------------------------------------------------------
// Fused attention: one block (8 waves) per 16 query rows.
//   Stage:  Q rows (16 x 1024, 32 KB) -> LDS via async global->LDS
//   Pass 1: S = Q @ K^T * 0.125 -> bf16 scores in 128 KB LDS (16 x 4096)
//   Softmax: per-row max/sum via 16-lane shuffles, P written in place
//   Pass 2: O = P @ V using pre-transposed V^T [1024 x 4096]
// qkv layout: [4096][3072] bf16, Q at col 0, K at 1024.
// Dynamic LDS = 160 KB (CDNA5 WGP has 320 KB).
// ---------------------------------------------------------------------------
__global__ __launch_bounds__(256)
void attention_kernel(const __bf16* __restrict__ qkv,
                      const __bf16* __restrict__ vT,
                      __bf16* __restrict__ attnout) {
  extern __shared__ char smem_raw[];
  __bf16* SP = (__bf16*)smem_raw;    // 16 x 4096 scores/probs
  __bf16* SQ = SP + 16 * 4096;       // 16 x 1024 staged Q

  const int tid = threadIdx.x;
  const int lane = tid & 31;
  const int wave = tid >> 5;
  const int qRow0 = blockIdx.x * 16;

  const __bf16* Km = qkv + 1024;
  const int ld = 3072;

  // ---- stage Q rows into LDS (async) ----
  for (int c = tid; c < 16 * 128; c += 256) {   // 16 rows x (1024/8) chunks
    int r  = c >> 7;
    int kk = (c & 127) << 3;
    stage_b128(qkv + (qRow0 + r) * ld + kk, SQ + r * 1024 + kk);
  }
  stage_wait();
  __syncthreads();

  // ---- Pass 1: scores (each wave handles 512 keys = 32 key tiles) ----
  for (int kt = 0; kt < 32; ++kt) {
    const int key0 = (wave * 32 + kt) * 16;
    v8f acc = {};
    for (int k0 = 0; k0 < 1024; k0 += 32) {
      v16bf a = ld_a_rowmajor(SQ, 1024, 0, k0, lane);      // LDS
      v16bf b = ld_bt_rowmajor(Km, ld, key0, k0, lane);    // global (L2)
      acc = wmma_bf16(a, b, acc);
    }
    const int rloc = (lane >> 4) << 3;
    const int c = key0 + (lane & 15);
#pragma unroll
    for (int v = 0; v < 8; ++v)
      SP[(rloc + v) * 4096 + c] = (__bf16)(acc[v] * 0.125f);
  }
  __syncthreads();

  // ---- Softmax: 16 threads per row; groups sit inside 16-lane halves ----
  {
    const int row = tid >> 4;
    const int sub = tid & 15;
    float m = -3.0e38f;
    for (int k = sub; k < 4096; k += 16)
      m = fmaxf(m, (float)SP[row * 4096 + k]);
#pragma unroll
    for (int msk = 1; msk < 16; msk <<= 1)
      m = fmaxf(m, __shfl_xor(m, msk, 32));

    float l = 0.0f;
    for (int k = sub; k < 4096; k += 16)
      l += __expf((float)SP[row * 4096 + k] - m);
#pragma unroll
    for (int msk = 1; msk < 16; msk <<= 1)
      l += __shfl_xor(l, msk, 32);

    const float inv = 1.0f / l;
    for (int k = sub; k < 4096; k += 16) {
      float p = __expf((float)SP[row * 4096 + k] - m) * inv;
      SP[row * 4096 + k] = (__bf16)p;
    }
  }
  __syncthreads();

  // ---- Pass 2: O = P @ V; wave owns output cols [wave*128, wave*128+128) ----
  const int col0 = wave * 128;
  v8f acc2[8] = {};
  for (int k0 = 0; k0 < 4096; k0 += 32) {
    v16bf a = ld_a_rowmajor(SP, 4096, 0, k0, lane);        // P from LDS
#pragma unroll
    for (int j = 0; j < 8; ++j) {
      v16bf b = ld_bt_rowmajor(vT, 4096, col0 + 16 * j, k0, lane);  // V^T
      acc2[j] = wmma_bf16(a, b, acc2[j]);
    }
  }
  const int rloc = (lane >> 4) << 3;
  const int cloc = lane & 15;
#pragma unroll
  for (int j = 0; j < 8; ++j)
#pragma unroll
    for (int v = 0; v < 8; ++v)
      attnout[(qRow0 + rloc + v) * 1024 + col0 + 16 * j + cloc] =
          (__bf16)acc2[j][v];
}

// ---------------------------------------------------------------------------
// Launcher
// ---------------------------------------------------------------------------
extern "C" void kernel_launch(void* const* d_in, const int* in_sizes, int n_in,
                              void* d_out, int out_size, void* d_ws, size_t ws_size,
                              hipStream_t stream) {
  (void)in_sizes; (void)n_in; (void)out_size; (void)ws_size;
  const float* x     = (const float*)d_in[0];  // 4096 x 1024
  const float* w_qkv = (const float*)d_in[1];  // 1024 x 3072
  const float* b_qkv = (const float*)d_in[2];  // 3072
  const float* w_out = (const float*)d_in[3];  // 1024 x 1024
  const float* b_out = (const float*)d_in[4];  // 1024
  float* out = (float*)d_out;                  // 4096 x 1024

  // Workspace carve-up (56 MB total)
  __bf16* xb     = (__bf16*)d_ws;                    // 4096x1024   8 MB
  __bf16* wqkvT  = xb     + (size_t)4096 * 1024;     // 3072x1024   6 MB
  __bf16* woutT  = wqkvT  + (size_t)3072 * 1024;     // 1024x1024   2 MB
  __bf16* qkvb   = woutT  + (size_t)1024 * 1024;     // 4096x3072  24 MB
  __bf16* vT     = qkvb   + (size_t)4096 * 3072;     // 1024x4096   8 MB
  __bf16* aob    = vT     + (size_t)1024 * 4096;     // 4096x1024   8 MB

  // x -> bf16 (no transpose needed)
  cvt_f32_bf16_kernel<<<(4096 * 1024 / 4 + 255) / 256, 256, 0, stream>>>(
      x, xb, 4096 * 1024 / 4);
  // weights: fused convert + transpose to N-major (B^T) for b128 fragments
  transpose_to_bf16_kernel<float><<<dim3(3072 / 32, 1024 / 32), dim3(32, 8),
                                    0, stream>>>(w_qkv, 3072, wqkvT, 1024);
  transpose_to_bf16_kernel<float><<<dim3(1024 / 32, 1024 / 32), dim3(32, 8),
                                    0, stream>>>(w_out, 1024, woutT, 1024);

  // qkv = x @ w_qkv + b_qkv   (M=4096, N=3072, K=1024), bf16 out
  gemm_bias_kernel<true><<<dim3(4096 / 16, 3072 / 512), 256, 0, stream>>>(
      xb, wqkvT, b_qkv, (void*)qkvb, 1024, 1024, 1024, 3072);

  // V (cols 2048..3071 of qkv) -> V^T [1024 x 4096]
  transpose_to_bf16_kernel<__bf16><<<dim3(1024 / 32, 4096 / 32), dim3(32, 8),
                                     0, stream>>>(qkvb + 2048, 3072, vT, 4096);

  // fused softmax(Q K^T / 8) @ V -> aob (bf16), 160 KB dynamic LDS
  attention_kernel<<<4096 / 16, 256, (16 * 4096 + 16 * 1024) * sizeof(__bf16),
                     stream>>>(qkvb, vT, aob);

  // out = aob @ w_out + b_out  (M=4096, N=1024, K=1024), f32 out
  gemm_bias_kernel<false><<<dim3(4096 / 16, 1024 / 512), 256, 0, stream>>>(
      aob, woutT, b_out, (void*)out, 1024, 1024, 1024, 1024);
}
